// DynamicConv_30039001268996
// MI455X (gfx1250) — matrix operand order repaired and checked by reading, not compiled, not run on hardware
//
#include <hip/hip_runtime.h>

typedef __attribute__((ext_vector_type(16))) __bf16        v16bf;
typedef __attribute__((ext_vector_type(8)))  float         v8f;
typedef __attribute__((ext_vector_type(4)))  float         v4f;
typedef __attribute__((ext_vector_type(4)))  unsigned int  v4u;

#define B_     16
#define C_IN   128
#define C_OUT  128
#define F_     64
#define T_     256
#define KK     1152            // 128 ic * 9 taps
#define FLAT   147456          // 128 oc * 1152
#define PLANE  (F_ * T_)       // 16384

// padded transposed activations: [b][ffp(66)][ttp(258)][ic(128)] bf16
#define FP_    66
#define TP_    258
#define XTP_ELEMS ((size_t)B_ * FP_ * TP_ * C_IN)          // 34,873,344
#define XTP_BYTES (XTP_ELEMS * 2)                          // 69,746,688

// ===========================================================================
// Shared helper: attention logits -> softmax(logit/30) weights a0..a3
// ===========================================================================
__device__ __forceinline__ void softmax4(const float* __restrict__ g,
                                         const float* __restrict__ dk,
                                         const float* __restrict__ dbias,
                                         int b, float* __restrict__ logit_s,
                                         float& a0, float& a1, float& a2, float& a3)
{
    if (threadIdx.x < 4) {
        float l = dbias[threadIdx.x];
        for (int c = 0; c < 128; ++c)
            l += g[b * 128 + c] * dk[c * 4 + threadIdx.x];
        logit_s[threadIdx.x] = l * (1.0f / 30.0f);
    }
    __syncthreads();
    const float l0 = logit_s[0], l1 = logit_s[1], l2 = logit_s[2], l3 = logit_s[3];
    const float m  = fmaxf(fmaxf(l0, l1), fmaxf(l2, l3));
    const float e0 = __expf(l0 - m), e1 = __expf(l1 - m),
                e2 = __expf(l2 - m), e3 = __expf(l3 - m);
    const float inv = 1.0f / (e0 + e1 + e2 + e3);
    a0 = e0 * inv; a1 = e1 * inv; a2 = e2 * inv; a3 = e3 * inv;
}

// ===========================================================================
// FAST PATH
// ===========================================================================

// --- zero-fill padded xt buffer (covers the halo; interior overwritten) -----
__global__ __launch_bounds__(256)
void zero_kernel(v4u* __restrict__ p, long n16)
{
    const long idx = (long)blockIdx.x * 256 + threadIdx.x;
    if (idx < n16) {
        v4u z = {0u, 0u, 0u, 0u};
        p[idx] = z;
    }
}

// --- x: NCHW f32 -> padded [b][ff+1][tt+1][ic] bf16 (LDS tile transpose) ----
__global__ __launch_bounds__(256)
void xpose_kernel(const float* __restrict__ x, __bf16* __restrict__ xt)
{
    const int b  = blockIdx.z;
    const int f  = blockIdx.y;
    const int t0 = blockIdx.x * 64;
    __shared__ float tile[128][65];          // pad -> conflict-free transpose

    const float* xp = x + ((size_t)b * C_IN * F_ + f) * T_ + t0;   // + ic*PLANE
#pragma unroll 4
    for (int it = 0; it < 32; ++it) {
        const int idx = it * 256 + threadIdx.x;
        const int ic = idx >> 6;             // 0..127
        const int tl = idx & 63;             // 0..63 (coalesced reads)
        tile[ic][tl] = xp[(size_t)ic * PLANE + tl];
    }
    __syncthreads();

    // padded destination: ffp = f+1, ttp = t0+1+tl
    __bf16* op = xt + (((size_t)b * FP_ + (f + 1)) * TP_ + (t0 + 1)) * C_IN;
#pragma unroll 4
    for (int it = 0; it < 32; ++it) {
        const int idx = it * 256 + threadIdx.x;
        const int tl = idx >> 7;             // 0..63
        const int ic = idx & 127;            // coalesced bf16 writes
        op[(size_t)tl * C_IN + ic] = (__bf16)tile[ic][tl];
    }
}

// --- attention + kernel aggregation, bf16, pre-swizzled A-fragment layout ---
// out layout: [b][oc_tile(8)][chunk(36)][lane(32)][elem(16)] bf16
// elem i of lane -> K = chunk*32 + (lane>>4)*16 + i
// K = r*128 + ic ; input weight flat index = oc*1152 + ic*9 + r
__global__ __launch_bounds__(256)
void agg_bf16_kernel(const float* __restrict__ g,
                     const float* __restrict__ dk,
                     const float* __restrict__ dbias,
                     const float* __restrict__ weight,
                     __bf16* __restrict__ aggB)
{
    const int b = blockIdx.y;
    __shared__ float logit_s[4];
    float a0, a1, a2, a3;
    softmax4(g, dk, dbias, b, logit_s, a0, a1, a2, a3);

    const int base = blockIdx.x * 1024;
#pragma unroll
    for (int e = 0; e < 4; ++e) {
        const int idx    = base + e * 256 + threadIdx.x;   // [0, FLAT)
        const int octile = idx / 18432;                    // 8*36*512 per b
        const int rem    = idx - octile * 18432;
        const int chunk  = rem >> 9;                       // 0..35
        const int rem2   = rem & 511;
        const int lane   = rem2 >> 4;
        const int i      = rem2 & 15;
        const int oc     = octile * 16 + (lane & 15);
        const int hi     = lane >> 4;
        const int K      = chunk * 32 + hi * 16 + i;
        const int ic     = K & 127;
        const int r      = K >> 7;
        const int inIdx  = oc * 1152 + ic * 9 + r;
        const float v = a0 * weight[inIdx]
                      + a1 * weight[FLAT + inIdx]
                      + a2 * weight[2 * FLAT + inIdx]
                      + a3 * weight[3 * FLAT + inIdx];
        aggB[(size_t)b * FLAT + idx] = (__bf16)v;          // coalesced store
    }
}

// --- conv: unconditional v16bf loads at immediate offsets, 36 WMMAs ---------
__global__ __launch_bounds__(128)
void conv_bf16_wmma(const __bf16* __restrict__ xt,    // padded [16][66][258][128]
                    const __bf16* __restrict__ aggB,  // swizzled
                    float* __restrict__ out)          // [16,128,64,256] f32
{
    const int b    = blockIdx.z;
    const int oct  = blockIdx.y;            // oc tile 0..7
    const int f    = blockIdx.x >> 2;
    const int t0   = (blockIdx.x & 3) * 64;
    const int lane = threadIdx.x & 31;
    const int wave = threadIdx.x >> 5;
    const int n    = lane & 15;
    const int hi   = lane >> 4;
    const int t    = t0 + wave * 16 + n;

    // A base: all 36 chunk offsets are immediates (chunk*1024 bytes)
    const __bf16* aRow = aggB + (((size_t)(b * 8 + oct) * 36) * 32 + lane) * 16;
    // B base: padded indexing -> tap (r) offsets are immediates
    const __bf16* xbase = xt + (((size_t)b * FP_ + f) * TP_ + t) * C_IN + hi * 16;

    v8f acc = {};
#pragma unroll
    for (int r = 0; r < 9; ++r) {
        const __bf16* xrow = xbase + (size_t)((r / 3) * TP_ + (r % 3)) * C_IN;
#pragma unroll
        for (int cc = 0; cc < 4; ++cc) {
            const v16bf A  = *(const v16bf*)(aRow + (size_t)(r * 4 + cc) * 512);
            const v16bf Bv = *(const v16bf*)(xrow + cc * 32);
            acc = __builtin_amdgcn_wmma_f32_16x16x32_bf16(
                false, A, false, Bv, (short)0, acc, false, false);
        }
    }

    float* outp = out + (((size_t)b * C_OUT + oct * 16 + hi * 8) * F_ + f) * T_ + t;
#pragma unroll
    for (int rr = 0; rr < 8; ++rr)
        outp[(size_t)rr * PLANE] = acc[rr];
}

// ===========================================================================
// FALLBACK PATH (small workspace): f32 agg [b][oc][r*128+ic], on-the-fly cvt
// ===========================================================================
__global__ __launch_bounds__(256)
void fb_agg_kernel(const float* __restrict__ g, const float* __restrict__ dk,
                   const float* __restrict__ dbias, const float* __restrict__ weight,
                   float* __restrict__ aggP)
{
    const int b = blockIdx.y;
    __shared__ float logit_s[4];
    float a0, a1, a2, a3;
    softmax4(g, dk, dbias, b, logit_s, a0, a1, a2, a3);

    const int base = blockIdx.x * 1024;
#pragma unroll
    for (int e = 0; e < 4; ++e) {
        const int outIdx = base + e * 256 + threadIdx.x;   // oc*1152 + r*128 + ic
        const int oc  = outIdx / 1152;
        const int rem = outIdx - oc * 1152;
        const int r   = rem >> 7;
        const int ic  = rem & 127;
        const int inIdx = oc * 1152 + ic * 9 + r;
        aggP[(size_t)b * FLAT + outIdx] =
              a0 * weight[inIdx] + a1 * weight[FLAT + inIdx]
            + a2 * weight[2 * FLAT + inIdx] + a3 * weight[3 * FLAT + inIdx];
    }
}

__global__ __launch_bounds__(128)
void fb_conv_wmma(const float* __restrict__ x, const float* __restrict__ aggP,
                  float* __restrict__ out)
{
    const int b    = blockIdx.z;
    const int oc0  = blockIdx.y * 16;
    const int f    = blockIdx.x >> 2;
    const int t0   = (blockIdx.x & 3) * 64;
    const int lane = threadIdx.x & 31;
    const int wave = threadIdx.x >> 5;
    const int n    = lane & 15;
    const int hi   = lane >> 4;
    const int t    = t0 + wave * 16 + n;
    const int oc   = oc0 + n;

    const float* aggRow = aggP + (size_t)b * FLAT + (size_t)oc * KK + hi * 8;
    const float* xb     = x + (size_t)b * C_IN * PLANE;

    v8f acc = {};
#pragma unroll
    for (int r = 0; r < 9; ++r) {
        const int ff = f + r / 3 - 1;
        const int tt = t + r % 3 - 1;
        const bool valid = ((unsigned)ff < (unsigned)F_) & ((unsigned)tt < (unsigned)T_);
        const float* xrow = xb + (ptrdiff_t)ff * T_ + tt;
#pragma unroll
        for (int cc = 0; cc < 4; ++cc) {
            const int k0 = r * 128 + cc * 32;
            float af[16];
            *(v4f*)(af + 0)  = *(const v4f*)(aggRow + k0);
            *(v4f*)(af + 4)  = *(const v4f*)(aggRow + k0 + 4);
            *(v4f*)(af + 8)  = *(const v4f*)(aggRow + k0 + 16);
            *(v4f*)(af + 12) = *(const v4f*)(aggRow + k0 + 20);
            v16bf A;
#pragma unroll
            for (int i = 0; i < 16; ++i) A[i] = (__bf16)af[i];

            const float* xcol = xrow + (size_t)(cc * 32 + hi * 8) * PLANE;
            v16bf Bv;
#pragma unroll
            for (int j = 0; j < 16; ++j)
                Bv[j] = (__bf16)(valid ? xcol[(size_t)(j < 8 ? j : j + 8) * PLANE] : 0.0f);

            acc = __builtin_amdgcn_wmma_f32_16x16x32_bf16(
                false, A, false, Bv, (short)0, acc, false, false);
        }
    }

    float* outp = out + (((size_t)b * C_OUT + oc0 + hi * 8) * F_ + f) * T_ + t;
#pragma unroll
    for (int rr = 0; rr < 8; ++rr)
        outp[(size_t)rr * PLANE] = acc[rr];
}

// ===========================================================================
extern "C" void kernel_launch(void* const* d_in, const int* in_sizes, int n_in,
                              void* d_out, int out_size, void* d_ws, size_t ws_size,
                              hipStream_t stream) {
    const float* x  = (const float*)d_in[0];   // [16,128,64,256]
    const float* g  = (const float*)d_in[1];   // [1,16,128]
    const float* dk = (const float*)d_in[2];   // [128,4]
    const float* db = (const float*)d_in[3];   // [4]
    const float* w  = (const float*)d_in[4];   // [1,1,4,147456]
    float* out = (float*)d_out;

    const size_t AGGB_BYTES = (size_t)B_ * FLAT * sizeof(__bf16);   // 4.7 MB

    if (ws_size >= AGGB_BYTES + XTP_BYTES) {
        __bf16* aggB = (__bf16*)d_ws;
        __bf16* xt   = (__bf16*)((char*)d_ws + AGGB_BYTES);

        const long n16 = (long)(XTP_BYTES / 16);
        zero_kernel   <<<dim3((unsigned)((n16 + 255) / 256)), 256, 0, stream>>>((v4u*)xt, n16);
        agg_bf16_kernel<<<dim3(144, B_), 256, 0, stream>>>(g, dk, db, w, aggB);
        xpose_kernel  <<<dim3(4, F_, B_), 256, 0, stream>>>(x, xt);
        conv_bf16_wmma<<<dim3(256, 8, B_), 128, 0, stream>>>(xt, aggB, out);
    } else {
        float* aggP = (float*)d_ws;                                 // 9.4 MB
        fb_agg_kernel<<<dim3(144, B_), 256, 0, stream>>>(g, dk, db, w, aggP);
        fb_conv_wmma <<<dim3(256, 8, B_), 128, 0, stream>>>(x, aggP, out);
    }
}
